// ChannelAttention_36953898614882
// MI455X (gfx1250) — matrix operand order repaired and checked by reading, not compiled, not run on hardware
//
#include <hip/hip_runtime.h>
#include <hip/hip_bf16.h>

typedef __bf16 bf16_t;
typedef __attribute__((ext_vector_type(16))) __bf16 v16bf;
typedef __attribute__((ext_vector_type(8)))  float  v8f;

#define NTOK 13824    // 24*24*24
#define BDIM 4
#define CDIM 256
#define NHEAD 8
#define DHEAD 32

union frag_t { v16bf v; unsigned u[8]; };

// ---------------------------------------------------------------------------
// Kernel 1: qkv projection GEMM.  out[m, j] = sum_c x[b, c, n] * W[c, j]
// m = b*NTOK + n (tile of 128 tokens never crosses b since 128 | NTOK),
// j in [0, 768) -> which = j>>8 (0=q,1=k,2=v), h = (j&255)>>5, dd = j&31.
// q,k stored bf16 [which][b*8+h][tok][32]; v stored fp32 [b*8+h][tok][32].
// ---------------------------------------------------------------------------
__global__ __launch_bounds__(256) void gemm_qkv_kernel(
    const float* __restrict__ x, const float* __restrict__ Wm,
    bf16_t* __restrict__ qk, float* __restrict__ vw)
{
    __shared__ __align__(16) bf16_t Alds[128 * 34]; // [m][kk], stride 34 (pad)
    __shared__ __align__(16) bf16_t Blds[128 * 34]; // [col][kk] transposed, stride 34

    const int tile = blockIdx.x;
    const int tn = tile % 6;          // 768/128 column tiles
    const int tm = tile / 6;
    const int m0 = tm * 128;
    const int b  = m0 / NTOK;
    const int n0 = m0 - b * NTOK;
    const int j0 = tn * 128;

    const int t    = threadIdx.x;
    const int wave = t >> 5;
    const int lane = t & 31;
    const int rlo  = lane & 15;
    const int hi   = lane >> 4;

    v8f zero = {0.f, 0.f, 0.f, 0.f, 0.f, 0.f, 0.f, 0.f};
    v8f acc[8];
#pragma unroll
    for (int j = 0; j < 8; ++j) acc[j] = zero;

    const unsigned* Au = (const unsigned*)Alds;
    const unsigned* Bu = (const unsigned*)Blds;

    for (int k0 = 0; k0 < CDIM; k0 += 32) {
        // Prefetch next K-tile lines (x fits in 192MB L2; hides k-loop latency).
        if (k0 + 32 < CDIM) {
            const int pcc  = t >> 5;
            const int pmm4 = (t & 31) * 4;
            __builtin_prefetch(&x[(size_t)(b * CDIM + k0 + 32 + pcc) * NTOK + n0 + pmm4], 0, 0);
            __builtin_prefetch(&Wm[(size_t)(k0 + 32 + pcc) * 1024 + j0 + pmm4], 0, 0);
        }
        // Stage A (x^T tile, 128 tokens x 32 channels) and B (W tile,
        // transposed to [col][k]) into LDS as bf16.  float4 global reads:
        // a wave covers 512B contiguous per issue.
#pragma unroll
        for (int i = 0; i < 4; ++i) {
            const int f4  = i * 256 + t;      // 0..1023 float4s
            const int cc  = f4 >> 5;          // k within 32-chunk
            const int mm4 = (f4 & 31) * 4;    // token (A) / col (B), step 4
            const float4 xa =
                *(const float4*)&x[(size_t)(b * CDIM + k0 + cc) * NTOK + n0 + mm4];
            Alds[(mm4 + 0) * 34 + cc] = (bf16_t)xa.x;
            Alds[(mm4 + 1) * 34 + cc] = (bf16_t)xa.y;
            Alds[(mm4 + 2) * 34 + cc] = (bf16_t)xa.z;
            Alds[(mm4 + 3) * 34 + cc] = (bf16_t)xa.w;
            const float4 wb =
                *(const float4*)&Wm[(size_t)(k0 + cc) * 1024 + j0 + mm4];
            Blds[(mm4 + 0) * 34 + cc] = (bf16_t)wb.x;
            Blds[(mm4 + 1) * 34 + cc] = (bf16_t)wb.y;
            Blds[(mm4 + 2) * 34 + cc] = (bf16_t)wb.z;
            Blds[(mm4 + 3) * 34 + cc] = (bf16_t)wb.w;
        }
        __syncthreads();

        // A fragment: lane<16 -> M=lane, K={0..7,16..23}; lane>=16 -> K+8.
        frag_t af;
        const int arow = wave * 16 + rlo;
#pragma unroll
        for (int r = 0; r < 4; ++r) {
            af.u[r]     = Au[arow * 17 + hi * 4 + r];
            af.u[4 + r] = Au[arow * 17 + 8 + hi * 4 + r];
        }
        // Materialize all 8 B fragments first so the scheduler can batch the
        // ds_loads and issue the 8 WMMAs back-to-back (no per-wmma dscnt=0).
        frag_t bfr[8];
#pragma unroll
        for (int j = 0; j < 8; ++j) {
            const int col = j * 16 + rlo;
#pragma unroll
            for (int r = 0; r < 8; ++r)
                bfr[j].u[r] = Bu[col * 17 + (r >> 2) * 8 + hi * 4 + (r & 3)];
        }
#pragma unroll
        for (int j = 0; j < 8; ++j)
            acc[j] = __builtin_amdgcn_wmma_f32_16x16x32_bf16(
                false, af.v, false, bfr[j].v, (short)0, acc[j], false, false);
        __syncthreads();
    }

    // Epilogue: C/D layout -> element (m = r + hi*8, n = rlo) in vgpr r.
#pragma unroll
    for (int j = 0; j < 8; ++j) {
        const int jg    = j0 + j * 16 + rlo;
        const int which = jg >> 8;            // uniform per j across wave
        const int rem   = jg & 255;
        const int h     = rem >> 5;
        const int dd    = rem & 31;
#pragma unroll
        for (int r = 0; r < 8; ++r) {
            const int tok = n0 + wave * 16 + r + hi * 8;
            const float val = acc[j][r];
            if (which < 2) {
                qk[((size_t)(which * 32 + b * NHEAD + h) * NTOK + tok) * 32 + dd] =
                    (bf16_t)val;
            } else {
                vw[((size_t)(b * NHEAD + h) * NTOK + tok) * 32 + dd] = val;
            }
        }
    }
}

// ---------------------------------------------------------------------------
// Kernel 2: per-(which,b,h,d) L2 norms over N of the bf16 q/k columns.
// grid = 64 blocks (which*32 + b*8 + h), 256 threads.
// ---------------------------------------------------------------------------
__global__ __launch_bounds__(256) void norms_kernel(
    const bf16_t* __restrict__ qk, float* __restrict__ norms)
{
    const int blk = blockIdx.x;                    // 0..63
    const bf16_t* src = qk + (size_t)blk * NTOK * 32;
    const int t = threadIdx.x;
    const int d = t & 31;
    const int g = t >> 5;
    float s = 0.f;
    for (int n = g; n < NTOK; n += 8) {
        float v = (float)src[(size_t)n * 32 + d];
        s += v * v;
    }
    __shared__ float red[256];
    red[t] = s;
    __syncthreads();
    if (t < 32) {
        float tot = 0.f;
#pragma unroll
        for (int gg = 0; gg < 8; ++gg) tot += red[gg * 32 + d];
        norms[blk * 32 + d] = fmaxf(sqrtf(tot), 1e-12f);
    }
}

// ---------------------------------------------------------------------------
// Kernel 3: per-(b,h): S = q^T k over N via WMMA, scale by norms & temp,
// row softmax -> attn[32][32] fp32.  grid = 32 blocks, 256 threads (8 waves).
// ---------------------------------------------------------------------------
__global__ __launch_bounds__(256) void gram_softmax_kernel(
    const bf16_t* __restrict__ qk, const float* __restrict__ norms,
    const float* __restrict__ temp, float* __restrict__ attn)
{
    const int bh   = blockIdx.x;   // b*8+h
    const int t    = threadIdx.x;
    const int wave = t >> 5;
    const int lane = t & 31;
    const int rlo  = lane & 15;
    const int hi   = lane >> 4;

    __shared__ __align__(16) float smem[8960];   // staging, reused for reduce
    __shared__ float Sf[1024];

    bf16_t* stq = (bf16_t*)smem + wave * (2 * 32 * 34);
    bf16_t* stk = stq + 32 * 34;
    const unsigned* Qu = (const unsigned*)stq;
    const unsigned* Ku = (const unsigned*)stk;

    const int per_wave = NTOK / 8;               // 1728
    const int tokbase  = wave * per_wave;

    const uint4* qg4 = (const uint4*)(qk + ((size_t)bh * NTOK) * 32);
    const uint4* kg4 = (const uint4*)(qk + ((size_t)(32 + bh) * NTOK) * 32);

    v8f zero = {0.f, 0.f, 0.f, 0.f, 0.f, 0.f, 0.f, 0.f};
    v8f acc[4];
#pragma unroll
    for (int s = 0; s < 4; ++s) acc[s] = zero;

    for (int c = 0; c < per_wave; c += 32) {
        const int tok0 = tokbase + c;
        // Stage 32 tokens x 32 d of q and k, transposed to [d][tok] (pad 34).
        // uint4 global reads: 16 uints per token => tok = u4>>2.
#pragma unroll
        for (int i = 0; i < 4; ++i) {
            const int u4  = i * 32 + lane;    // 0..127
            const int tok = u4 >> 2;
            const int p   = (u4 & 3) * 4;     // uint offset within token row
            union { uint4 v; unsigned u[4]; } cq, ck;
            cq.v = qg4[tok0 * 4 + u4];
            ck.v = kg4[tok0 * 4 + u4];
#pragma unroll
            for (int w = 0; w < 4; ++w) {
                union { unsigned u; bf16_t h[2]; } tq, tk;
                tq.u = cq.u[w];
                tk.u = ck.u[w];
                const int dd = (p + w) * 2;
                stq[dd * 34 + tok]       = tq.h[0];
                stq[(dd + 1) * 34 + tok] = tq.h[1];
                stk[dd * 34 + tok]       = tk.h[0];
                stk[(dd + 1) * 34 + tok] = tk.h[1];
            }
        }
        __syncthreads();

        frag_t a0, a1, b0, b1;
#pragma unroll
        for (int r = 0; r < 4; ++r) {
            a0.u[r]     = Qu[rlo * 17 + hi * 4 + r];
            a0.u[4 + r] = Qu[rlo * 17 + 8 + hi * 4 + r];
            a1.u[r]     = Qu[(16 + rlo) * 17 + hi * 4 + r];
            a1.u[4 + r] = Qu[(16 + rlo) * 17 + 8 + hi * 4 + r];
        }
#pragma unroll
        for (int r = 0; r < 8; ++r) {
            b0.u[r] = Ku[rlo * 17 + (r >> 2) * 8 + hi * 4 + (r & 3)];
            b1.u[r] = Ku[(16 + rlo) * 17 + (r >> 2) * 8 + hi * 4 + (r & 3)];
        }
        acc[0] = __builtin_amdgcn_wmma_f32_16x16x32_bf16(
            false, a0.v, false, b0.v, (short)0, acc[0], false, false);
        acc[1] = __builtin_amdgcn_wmma_f32_16x16x32_bf16(
            false, a0.v, false, b1.v, (short)0, acc[1], false, false);
        acc[2] = __builtin_amdgcn_wmma_f32_16x16x32_bf16(
            false, a1.v, false, b0.v, (short)0, acc[2], false, false);
        acc[3] = __builtin_amdgcn_wmma_f32_16x16x32_bf16(
            false, a1.v, false, b1.v, (short)0, acc[3], false, false);
        __syncthreads();
    }

    // Cross-wave reduction of partial S tiles.
    float* Sred = smem;   // [wave][sub][r][lane] = 8*4*8*32 floats
#pragma unroll
    for (int sub = 0; sub < 4; ++sub)
#pragma unroll
        for (int r = 0; r < 8; ++r)
            Sred[((wave * 4 + sub) * 8 + r) * 32 + lane] = acc[sub][r];
    __syncthreads();

    for (int s = t; s < 1024; s += 256) {
        const int d = s >> 5, e = s & 31;
        const int sub = (d >> 4) * 2 + (e >> 4);
        const int r   = d & 7;
        const int ln  = ((d >> 3) & 1) * 16 + (e & 15);
        float sum = 0.f;
#pragma unroll
        for (int w = 0; w < 8; ++w)
            sum += Sred[((w * 4 + sub) * 8 + r) * 32 + ln];
        Sf[s] = sum;
    }
    __syncthreads();

    if (t < 32) {
        const int d = t;
        const float nq = norms[bh * 32 + d];
        const float tp = temp[bh & 7];
        float row[32];
        float mx = -1e30f;
#pragma unroll
        for (int e = 0; e < 32; ++e) {
            const float nk = norms[1024 + bh * 32 + e];
            const float v = Sf[d * 32 + e] / (nq * nk) * tp;
            row[e] = v;
            mx = fmaxf(mx, v);
        }
        float ssum = 0.f;
#pragma unroll
        for (int e = 0; e < 32; ++e) {
            row[e] = expf(row[e] - mx);
            ssum += row[e];
        }
        const float inv = 1.f / ssum;
#pragma unroll
        for (int e = 0; e < 32; ++e)
            attn[bh * 1024 + d * 32 + e] = row[e] * inv;
    }
}

// ---------------------------------------------------------------------------
// Kernel 4: out[b, h*32+d, n] = sum_e attn[bh][d][e] * v[bh][n][e]
// grid = (NTOK/256, 32); one thread per token; bandwidth bound.
// ---------------------------------------------------------------------------
__global__ __launch_bounds__(256) void attn_v_kernel(
    const float* __restrict__ attn, const float* __restrict__ vw,
    float* __restrict__ out)
{
    const int bh = blockIdx.y;
    const int b  = bh >> 3;
    const int h  = bh & 7;
    const int n  = blockIdx.x * 256 + threadIdx.x;

    __shared__ float A[1024];
    for (int i = threadIdx.x; i < 1024; i += 256) A[i] = attn[bh * 1024 + i];
    __syncthreads();

    const float4* vp = (const float4*)(vw + ((size_t)bh * NTOK + n) * 32);
    float vv[32];
#pragma unroll
    for (int i = 0; i < 8; ++i) {
        float4 f = vp[i];
        vv[4 * i + 0] = f.x; vv[4 * i + 1] = f.y;
        vv[4 * i + 2] = f.z; vv[4 * i + 3] = f.w;
    }
    float* op = out + (size_t)(b * CDIM + h * DHEAD) * NTOK + n;
#pragma unroll
    for (int d = 0; d < 32; ++d) {
        float s = 0.f;
#pragma unroll
        for (int e = 0; e < 32; ++e) s += A[d * 32 + e] * vv[e];
        op[(size_t)d * NTOK] = s;
    }
}

// ---------------------------------------------------------------------------
extern "C" void kernel_launch(void* const* d_in, const int* in_sizes, int n_in,
                              void* d_out, int out_size, void* d_ws, size_t ws_size,
                              hipStream_t stream)
{
    (void)in_sizes; (void)n_in; (void)out_size; (void)ws_size;
    const float* x    = (const float*)d_in[0];
    const float* Wm   = (const float*)d_in[1];
    const float* temp = (const float*)d_in[2];
    float* out = (float*)d_out;

    // Workspace carve-up (all 16B aligned):
    //   qk  : bf16 [2][32][NTOK][32]  = 56,623,104 B
    //   vw  : f32  [32][NTOK][32]     = 56,623,104 B
    //   nrm : f32  [64][32]           = 8,192 B
    //   attn: f32  [32][32][32]       = 131,072 B
    bf16_t* qk  = (bf16_t*)d_ws;
    float*  vw  = (float*)((char*)d_ws + (size_t)2 * 32 * NTOK * 32 * sizeof(bf16_t));
    float*  nrm = vw + (size_t)32 * NTOK * 32;
    float*  att = nrm + 2048;

    const int m_tiles = (BDIM * NTOK) / 128;   // 432
    gemm_qkv_kernel<<<m_tiles * 6, 256, 0, stream>>>(x, Wm, qk, vw);
    norms_kernel<<<64, 256, 0, stream>>>(qk, nrm);
    gram_softmax_kernel<<<32, 256, 0, stream>>>(qk, nrm, temp, att);
    attn_v_kernel<<<dim3(NTOK / 256, 32), 256, 0, stream>>>(att, vw, out);
}
